// BahdanauAttention_66125316489781
// MI455X (gfx1250) — matrix operand order repaired and checked by reading, not compiled, not run on hardware
//
#include <hip/hip_runtime.h>
#include <cstdint>
#include <cstddef>

// Problem constants
constexpr int Bc = 512, Lc = 64, Fc = 2048, Hc = 512, Uc = 512;
constexpr int Mc = Bc * Lc;          // 32768 rows of the big GEMM
constexpr int KC = Fc + Hc;          // 2560, K of final GEMM

typedef __attribute__((ext_vector_type(16))) __bf16 v16bf;
typedef __attribute__((ext_vector_type(8)))  float  v8f;

// ---- gfx1250 async global->LDS path (guarded; falls back to sync copies) ---
#if __has_builtin(__builtin_amdgcn_global_load_async_to_lds_b128)
#define HAVE_ASYNC_B128 1
#else
#define HAVE_ASYNC_B128 0
#endif
#if __has_builtin(__builtin_amdgcn_global_load_async_to_lds_b64)
#define HAVE_ASYNC_B64 1
#else
#define HAVE_ASYNC_B64 0
#endif

// Exact pointee types per hipcc diagnostic:
//   param0: int __attribute__((vector_size(16))) __device__(AS1) *
//   param1: LDS-side pointer (AS3)
typedef int v4i_t __attribute__((vector_size(16)));
typedef int v2i_t __attribute__((vector_size(8)));
#define GAS_P(T, p) ((__attribute__((address_space(1))) T*)(p))
#define LAS_P(T, p) ((__attribute__((address_space(3))) T*)(p))

__device__ __forceinline__ void async_wait0() {
#if HAVE_ASYNC_B128 || HAVE_ASYNC_B64
#if __has_builtin(__builtin_amdgcn_s_wait_asynccnt)
  __builtin_amdgcn_s_wait_asynccnt(0);
#else
  asm volatile("s_wait_asynccnt 0x0" ::: "memory");
#endif
#endif
}

__device__ __forceinline__ __bf16 f2bf(float f) {
  union { float f; uint32_t u; } x; x.f = f;
  uint32_t r = x.u + 0x7FFFu + ((x.u >> 16) & 1u);  // round-to-nearest-even
  union { uint16_t s; __bf16 b; } y; y.s = (uint16_t)(r >> 16);
  return y.b;
}

// ---------------------------------------------------------------------------
// Convert + transpose f32 (K x N, row-major) -> bf16 (N x K, row-major)
// ---------------------------------------------------------------------------
__global__ void convtrans_kernel(const float* __restrict__ in, __bf16* __restrict__ out,
                                 int Kdim, int Ndim) {
  int idx = blockIdx.x * blockDim.x + threadIdx.x;
  if (idx >= Kdim * Ndim) return;
  int n = idx / Kdim;
  int k = idx - n * Kdim;
  out[(size_t)n * Kdim + k] = f2bf(in[(size_t)k * Ndim + n]);
}

// ---------------------------------------------------------------------------
// hW2[b,u] = hidden[b,:] @ W2[:,u] + b2[u]        (512 x 512 x 512, tiny)
// ---------------------------------------------------------------------------
__global__ void hw2_kernel(const float* __restrict__ hidden, const float* __restrict__ W2,
                           const float* __restrict__ b2, float* __restrict__ hW2) {
  __shared__ float hsh[Hc];
  int b = blockIdx.y;
  int u = blockIdx.x * 256 + threadIdx.x;
  for (int i = threadIdx.x; i < Hc; i += 256) hsh[i] = hidden[(size_t)b * Hc + i];
  __syncthreads();
  float acc = b2[u];
#pragma unroll 8
  for (int h = 0; h < Hc; ++h) acc += hsh[h] * W2[(size_t)h * Uc + u];
  hW2[(size_t)b * Uc + u] = acc;
}

// ---------------------------------------------------------------------------
// p{1,2}[b,l] = sigmoid( tanh(hidden[b,:] @ ka) @ kb )[l] * sqrt(L)
// ---------------------------------------------------------------------------
__global__ void localp_kernel(const float* __restrict__ hidden,
                              const float* __restrict__ ka1, const float* __restrict__ kb1,
                              const float* __restrict__ ka2, const float* __restrict__ kb2,
                              float* __restrict__ p1, float* __restrict__ p2) {
  __shared__ float hsh[Hc];
  __shared__ float s[Lc];
  int b = blockIdx.x, l = threadIdx.x;
  for (int i = l; i < Hc; i += Lc) hsh[i] = hidden[(size_t)b * Hc + i];
  __syncthreads();
  {
    float a = 0.f;
    for (int h = 0; h < Hc; ++h) a += hsh[h] * ka1[h * Lc + l];
    s[l] = tanhf(a);
    __syncthreads();
    float t = 0.f;
    for (int j = 0; j < Lc; ++j) t += s[j] * kb1[j * Lc + l];
    p1[b * Lc + l] = 8.0f / (1.0f + expf(-t));   // sqrt(64) = 8
    __syncthreads();
  }
  {
    float a = 0.f;
    for (int h = 0; h < Hc; ++h) a += hsh[h] * ka2[h * Lc + l];
    s[l] = tanhf(a);
    __syncthreads();
    float t = 0.f;
    for (int j = 0; j < Lc; ++j) t += s[j] * kb2[j * Lc + l];
    p2[b * Lc + l] = 8.0f / (1.0f + expf(-t));
    __syncthreads();
  }
}

// ---------------------------------------------------------------------------
// GEMM1 (WMMA bf16, double-buffered, async B-tile):
//   score = tanh(features@W1 + b1 + hW2[b]);
//   logits_part[g][m] = sum_{u in group g} score[m,u]*V[u]
// 512 threads (16 waves = 4 M-strips x 4 N-groups), 64-row tile == batch b,
// full N=512 resident in LDS so features stream from HBM exactly once.
// ---------------------------------------------------------------------------
constexpr int BKc = 32;
constexpr int STRc = BKc + 8;   // LDS stride in elements (conflict-free)
constexpr int NK1 = Fc / BKc;   // 64 K-steps
constexpr int NK2 = KC / BKc;   // 80 K-steps

__global__ void __launch_bounds__(512)
gemm1_wmma_kernel(const float*  __restrict__ features,  // (M, F) f32
                  const __bf16* __restrict__ W1t,       // (U, F) bf16 (pre-transposed)
                  const float*  __restrict__ b1,
                  const float*  __restrict__ Vv,
                  const float*  __restrict__ hW2,       // (B, U)
                  float*        __restrict__ logits_part) // (4, M)
{
  __shared__ __bf16 As[2][64 * STRc];     // 2 x  5 KB
  __shared__ __bf16 Bs[2][Uc * STRc];     // 2 x 40 KB   (Bs[n][k], transposed)

  const int tid   = threadIdx.x;
  const int lane  = tid & 31, wave = tid >> 5;
  const int mstrip = wave & 3, ngroup = wave >> 2;
  const int col16 = lane & 15, khalf = lane >> 4, koffA = khalf * 8;
  const int blockRow = blockIdx.x;               // == batch index b
  const size_t rowBase = (size_t)blockRow * 64;

  v8f acc[8] = {};

  const int ar = tid >> 3;           // 0..63
  const int ac = (tid & 7) * 4;      // 0..28
  const float*  aSrc = features + (rowBase + ar) * (size_t)Fc + ac;
  const __bf16* bSrc = W1t + (size_t)tid * Fc;

  auto loadB = [&](int k0, int buf) {
    const __bf16* s = bSrc + k0;
    __bf16* d = &Bs[buf][tid * STRc];
#if HAVE_ASYNC_B128
    __builtin_amdgcn_global_load_async_to_lds_b128(GAS_P(v4i_t, s), LAS_P(v4i_t, d), 0,  0);
    __builtin_amdgcn_global_load_async_to_lds_b128(GAS_P(v4i_t, s), LAS_P(v4i_t, d), 16, 0);
    __builtin_amdgcn_global_load_async_to_lds_b128(GAS_P(v4i_t, s), LAS_P(v4i_t, d), 32, 0);
    __builtin_amdgcn_global_load_async_to_lds_b128(GAS_P(v4i_t, s), LAS_P(v4i_t, d), 48, 0);
#else
    const uint4* s4 = (const uint4*)s;
    uint4* d4 = (uint4*)d;
    d4[0] = s4[0]; d4[1] = s4[1]; d4[2] = s4[2]; d4[3] = s4[3];
#endif
  };
  auto loadA = [&](int k0, int buf) {   // f32 -> bf16 conversion through VGPRs
    const float4 fv = *(const float4*)(aSrc + k0);
    union { __bf16 h[4]; uint2 v; } p;
    p.h[0] = f2bf(fv.x); p.h[1] = f2bf(fv.y); p.h[2] = f2bf(fv.z); p.h[3] = f2bf(fv.w);
    *(uint2*)(&As[buf][ar * STRc + ac]) = p.v;
  };

  loadB(0, 0);
  loadA(0, 0);
  async_wait0();
  __syncthreads();

  for (int ki = 0; ki < NK1; ++ki) {
    const int cur = ki & 1, nxt = cur ^ 1;
    if (ki + 1 < NK1) {               // prefetch next tile while computing
      loadB((ki + 1) * BKc, nxt);
      loadA((ki + 1) * BKc, nxt);
    }

    // A fragment (ISA 16-bit A layout: K in {koffA+0..7} then {16+koffA..})
    v16bf a;
    {
      uint32_t* ap = (uint32_t*)&a;
      const __bf16* Arow = &As[cur][(mstrip * 16 + col16) * STRc];
#pragma unroll
      for (int j = 0; j < 8; ++j) {
        int k = (j < 4) ? (koffA + 2 * j) : (16 + koffA + 2 * (j - 4));
        ap[j] = *(const uint32_t*)(Arow + k);
      }
    }
#pragma unroll
    for (int t = 0; t < 8; ++t) {
      v16bf bb;
      uint32_t* bp = (uint32_t*)&bb;
      const __bf16* Brow = &Bs[cur][(ngroup * 128 + t * 16 + col16) * STRc];
#pragma unroll
      for (int j = 0; j < 8; ++j) {          // B layout: same K order as A
        int k = (j < 4) ? (koffA + 2 * j) : (16 + koffA + 2 * (j - 4));
        bp[j] = *(const uint32_t*)(Brow + k);
      }
      acc[t] = __builtin_amdgcn_wmma_f32_16x16x32_bf16(
          false, a, false, bb, (short)0, acc[t], false, false);
    }

    async_wait0();
    __syncthreads();
  }

  // Epilogue: tanh + dot with V, reduce across the 16 lanes of each half
  const int mlocal = mstrip * 16 + (khalf ? 8 : 0);
  const float* hrow = hW2 + (size_t)blockRow * Uc;
  float part[8] = {};
#pragma unroll
  for (int t = 0; t < 8; ++t) {
    int u = ngroup * 128 + t * 16 + col16;
    float bias = b1[u] + hrow[u];
    float vv = Vv[u];
#pragma unroll
    for (int v = 0; v < 8; ++v)
      part[v] += tanhf(acc[t][v] + bias) * vv;
  }
#pragma unroll
  for (int v = 0; v < 8; ++v) {
    float p = part[v];
    p += __shfl_xor(p, 1, 32);
    p += __shfl_xor(p, 2, 32);
    p += __shfl_xor(p, 4, 32);
    p += __shfl_xor(p, 8, 32);
    if (col16 == 0)
      logits_part[(size_t)ngroup * Mc + rowBase + mlocal + v] = p;
  }
}

// ---------------------------------------------------------------------------
// softmax over L, then * exp(-2*d2)   (one 64-thread block per batch)
// bV omitted: softmax is shift-invariant and the mask doesn't use it.
// ---------------------------------------------------------------------------
__global__ void softmax_attn_kernel(const float* __restrict__ logits_part,
                                    const float* __restrict__ p1, const float* __restrict__ p2,
                                    float* __restrict__ attn_ws, float* __restrict__ attn_out) {
  __shared__ float sh[Lc];
  int b = blockIdx.x, l = threadIdx.x;
  size_t m = (size_t)b * Lc + l;
  float x = logits_part[m] + logits_part[Mc + m] + logits_part[2 * Mc + m] + logits_part[3 * Mc + m];
  sh[l] = x; __syncthreads();
  float mx = -1e30f;
  for (int j = 0; j < Lc; ++j) mx = fmaxf(mx, sh[j]);
  float e = expf(x - mx);
  __syncthreads(); sh[l] = e; __syncthreads();
  float sum = 0.f;
  for (int j = 0; j < Lc; ++j) sum += sh[j];
  float w = e / sum;
  float fl = (float)l;
  float rw = floorf((fl + 1.0f) * 0.125f);
  float cl = floorf(fmodf(fl + 1.0f, 8.0f)) - 1.0f;
  float d1 = rw - p1[m], d2 = cl - p2[m];
  w *= expf(-2.0f * (d1 * d1 + d2 * d2));
  attn_ws[m] = w;
  attn_out[m] = w;
}

// ---------------------------------------------------------------------------
// Tb[b, F+h] = bf16(tanh(hidden[b,h]))
// ---------------------------------------------------------------------------
__global__ void tanh_hidden_kernel(const float* __restrict__ hidden, __bf16* __restrict__ Tb) {
  int idx = blockIdx.x * 256 + threadIdx.x;   // over B*H
  int b = idx >> 9, h = idx & 511;
  Tb[(size_t)b * KC + Fc + h] = f2bf(tanhf(hidden[idx]));
}

// ---------------------------------------------------------------------------
// context[b,f] = sum_l attn[b,l]*features[b,l,f]; Tb[b,f] = bf16(tanh(ctx))
// ---------------------------------------------------------------------------
__global__ void context_kernel(const float* __restrict__ features,
                               const float* __restrict__ attn,
                               __bf16* __restrict__ Tb) {
  __shared__ float w[Lc];
  int b = blockIdx.y;
  int f = blockIdx.x * 256 + threadIdx.x;
  if (threadIdx.x < Lc) w[threadIdx.x] = attn[b * Lc + threadIdx.x];
  __syncthreads();
  const float* fb = features + (size_t)b * Lc * Fc + f;
  float acc = 0.f;
#pragma unroll 4
  for (int l = 0; l < Lc; ++l) {
    if (l < Lc - 2) __builtin_prefetch(fb + (size_t)(l + 2) * Fc, 0, 1); // global_prefetch_b8
    acc += w[l] * fb[(size_t)l * Fc];
  }
  Tb[(size_t)b * KC + f] = f2bf(tanhf(acc));
}

// ---------------------------------------------------------------------------
// GEMM2 (WMMA bf16, double-buffered, async A+B): out = Tb @ W3 + b3
// ---------------------------------------------------------------------------
__global__ void __launch_bounds__(512)
gemm2_wmma_kernel(const __bf16* __restrict__ Tb,       // (B, KC) bf16
                  const __bf16* __restrict__ W3t,      // (U, KC) bf16 (pre-transposed)
                  const float*  __restrict__ b3,
                  float*        __restrict__ out)      // (B, U)
{
  __shared__ __bf16 As[2][64 * STRc];
  __shared__ __bf16 Bs[2][Uc * STRc];

  const int tid = threadIdx.x;
  const int lane = tid & 31, wave = tid >> 5;
  const int mstrip = wave & 3, ngroup = wave >> 2;
  const int col16 = lane & 15, khalf = lane >> 4, koffA = khalf * 8;
  const size_t rowBase = (size_t)blockIdx.x * 64;

  v8f acc[8] = {};
  const int ar = tid >> 3;
  const int ac = (tid & 7) * 4;
  const __bf16* aSrc = Tb + (rowBase + ar) * (size_t)KC + ac;
  const __bf16* bSrc = W3t + (size_t)tid * KC;

  auto loadB = [&](int k0, int buf) {
    const __bf16* s = bSrc + k0;
    __bf16* d = &Bs[buf][tid * STRc];
#if HAVE_ASYNC_B128
    __builtin_amdgcn_global_load_async_to_lds_b128(GAS_P(v4i_t, s), LAS_P(v4i_t, d), 0,  0);
    __builtin_amdgcn_global_load_async_to_lds_b128(GAS_P(v4i_t, s), LAS_P(v4i_t, d), 16, 0);
    __builtin_amdgcn_global_load_async_to_lds_b128(GAS_P(v4i_t, s), LAS_P(v4i_t, d), 32, 0);
    __builtin_amdgcn_global_load_async_to_lds_b128(GAS_P(v4i_t, s), LAS_P(v4i_t, d), 48, 0);
#else
    const uint4* s4 = (const uint4*)s;
    uint4* d4 = (uint4*)d;
    d4[0] = s4[0]; d4[1] = s4[1]; d4[2] = s4[2]; d4[3] = s4[3];
#endif
  };
  auto loadA = [&](int k0, int buf) {
    const __bf16* s = aSrc + k0;
    __bf16* d = &As[buf][ar * STRc + ac];
#if HAVE_ASYNC_B64
    __builtin_amdgcn_global_load_async_to_lds_b64(GAS_P(v2i_t, s), LAS_P(v2i_t, d), 0, 0);
#else
    *(uint2*)d = *(const uint2*)s;
#endif
  };

  loadB(0, 0);
  loadA(0, 0);
  async_wait0();
  __syncthreads();

  for (int ki = 0; ki < NK2; ++ki) {
    const int cur = ki & 1, nxt = cur ^ 1;
    if (ki + 1 < NK2) {
      loadB((ki + 1) * BKc, nxt);
      loadA((ki + 1) * BKc, nxt);
    }

    v16bf a;
    {
      uint32_t* ap = (uint32_t*)&a;
      const __bf16* Arow = &As[cur][(mstrip * 16 + col16) * STRc];
#pragma unroll
      for (int j = 0; j < 8; ++j) {
        int k = (j < 4) ? (koffA + 2 * j) : (16 + koffA + 2 * (j - 4));
        ap[j] = *(const uint32_t*)(Arow + k);
      }
    }
#pragma unroll
    for (int t = 0; t < 8; ++t) {
      v16bf bb;
      uint32_t* bp = (uint32_t*)&bb;
      const __bf16* Brow = &Bs[cur][(ngroup * 128 + t * 16 + col16) * STRc];
#pragma unroll
      for (int j = 0; j < 8; ++j) {
        int k = (j < 4) ? (koffA + 2 * j) : (16 + koffA + 2 * (j - 4));
        bp[j] = *(const uint32_t*)(Brow + k);
      }
      acc[t] = __builtin_amdgcn_wmma_f32_16x16x32_bf16(
          false, a, false, bb, (short)0, acc[t], false, false);
    }

    async_wait0();
    __syncthreads();
  }

  const int mlocal = mstrip * 16 + (khalf ? 8 : 0);
#pragma unroll
  for (int t = 0; t < 8; ++t) {
    int u = ngroup * 128 + t * 16 + col16;
    float b3v = b3[u];
#pragma unroll
    for (int v = 0; v < 8; ++v)
      out[(rowBase + mlocal + v) * Uc + u] = acc[t][v] + b3v;
  }
}

// ---------------------------------------------------------------------------
extern "C" void kernel_launch(void* const* d_in, const int* in_sizes, int n_in,
                              void* d_out, int out_size, void* d_ws, size_t ws_size,
                              hipStream_t stream) {
  (void)in_sizes; (void)n_in; (void)out_size; (void)ws_size;
  const float* features = (const float*)d_in[0];
  const float* hidden   = (const float*)d_in[1];
  const float* W1  = (const float*)d_in[2];
  const float* b1  = (const float*)d_in[3];
  const float* W2  = (const float*)d_in[4];
  const float* b2  = (const float*)d_in[5];
  const float* Vv  = (const float*)d_in[6];
  // d_in[7] = bV: softmax is shift-invariant; bV cancels in both outputs.
  const float* W3  = (const float*)d_in[8];
  const float* b3  = (const float*)d_in[9];
  const float* ka1 = (const float*)d_in[10];
  const float* kb1 = (const float*)d_in[11];
  const float* ka2 = (const float*)d_in[12];
  const float* kb2 = (const float*)d_in[13];

  float* out      = (float*)d_out;                 // (B, U)
  float* attn_out = out + (size_t)Bc * Uc;         // (B, L, 1)

  // workspace bump allocator (~9 MiB total)
  char* wsp = (char*)d_ws;
  auto alloc = [&](size_t bytes) -> void* {
    void* p = (void*)wsp;
    wsp += (bytes + 255) & ~(size_t)255;
    return p;
  };
  __bf16* W1t        = (__bf16*)alloc((size_t)Uc * Fc * 2);    // 2   MiB
  __bf16* W3t        = (__bf16*)alloc((size_t)Uc * KC * 2);    // 2.5 MiB
  __bf16* Tb         = (__bf16*)alloc((size_t)Bc * KC * 2);    // 2.5 MiB
  float*  hW2        = (float*)alloc((size_t)Bc * Uc * 4);     // 1   MiB
  float*  p1         = (float*)alloc((size_t)Mc * 4);
  float*  p2         = (float*)alloc((size_t)Mc * 4);
  float*  logitsPart = (float*)alloc((size_t)4 * Mc * 4);      // 0.5 MiB
  float*  attn_ws    = (float*)alloc((size_t)Mc * 4);

  convtrans_kernel<<<(Uc * Fc + 255) / 256, 256, 0, stream>>>(W1, W1t, Fc, Uc);
  convtrans_kernel<<<(Uc * KC + 255) / 256, 256, 0, stream>>>(W3, W3t, KC, Uc);
  hw2_kernel<<<dim3(Uc / 256, Bc), 256, 0, stream>>>(hidden, W2, b2, hW2);
  localp_kernel<<<Bc, Lc, 0, stream>>>(hidden, ka1, kb1, ka2, kb2, p1, p2);
  gemm1_wmma_kernel<<<Mc / 64, 512, 0, stream>>>(features, W1t, b1, Vv, hW2, logitsPart);
  softmax_attn_kernel<<<Bc, Lc, 0, stream>>>(logitsPart, p1, p2, attn_ws, attn_out);
  tanh_hidden_kernel<<<(Bc * Hc) / 256, 256, 0, stream>>>(hidden, Tb);
  context_kernel<<<dim3(Fc / 256, Bc), 256, 0, stream>>>(features, attn_ws, Tb);
  gemm2_wmma_kernel<<<Bc / 64, 512, 0, stream>>>(Tb, W3t, b3, out);
}
